// VAE_45775761440871
// MI455X (gfx1250) — matrix operand order repaired and checked by reading, not compiled, not run on hardware
//
#include <hip/hip_runtime.h>

// ---------------------------------------------------------------------------
// CDNA5 (gfx1250) bf16-WMMA implementation of the LSTM-VAE reference.
//   E=512, H=1024, V=32000, B=64, T=64
// GEMMs: v_wmma_f32_16x16x32_bf16 fed from LDS, double-buffered with
// global_load_async_to_lds_b128 (ASYNCcnt) for DMA/compute overlap.
// All 4 B fragments + A fragment loaded before the WMMA chain so the four
// WMMAs issue back-to-back behind a single s_wait_dscnt.
// ---------------------------------------------------------------------------

typedef __bf16 bf16;
typedef __attribute__((ext_vector_type(16))) __bf16 v16bf;
typedef __attribute__((ext_vector_type(8)))  __bf16 bf16x8;
typedef __attribute__((ext_vector_type(8)))  float  v8f;

#define EDIM   512
#define HDIM   1024
#define VDIM   32000
#define BATCH  64
#define TSEQ   64

// ----------------------- CDNA5 async-to-LDS helpers -----------------------

__device__ __forceinline__ unsigned lds_off_of(const void* p) {
  // generic -> LDS address space; ptrtoint gives the LDS byte offset
  return (unsigned)(unsigned long long)(__attribute__((address_space(3))) const void*)p;
}

__device__ __forceinline__ void async_copy_b128(unsigned lds_byte_off, const void* gsrc) {
  // GLOBAL_LOAD_ASYNC_TO_LDS_B128: per-lane 16B global -> LDS, tracked by ASYNCcnt
  asm volatile("global_load_async_to_lds_b128 %0, %1, off"
               :: "v"(lds_byte_off), "v"(gsrc)
               : "memory");
}

__device__ __forceinline__ void wait_async0() {
  asm volatile("s_wait_asynccnt 0x0" ::: "memory");
}

// ------------------------- tiny elementwise kernels -------------------------

__global__ void k_f32_to_bf16(const float* __restrict__ in, bf16* __restrict__ out, int n) {
  int i = blockIdx.x * blockDim.x + threadIdx.x;
  if (i < n) out[i] = (bf16)in[i];
}

__global__ void k_bias_sum(const float* __restrict__ a, const float* __restrict__ b,
                           float* __restrict__ out, int n) {
  int i = blockIdx.x * blockDim.x + threadIdx.x;
  if (i < n) out[i] = a[i] + b[i];
}

__global__ void k_zero_f32(float* __restrict__ p, int n) {
  int i = blockIdx.x * blockDim.x + threadIdx.x;
  if (i < n) p[i] = 0.0f;
}

__global__ void k_zero_bf16(bf16* __restrict__ p, int n) {
  int i = blockIdx.x * blockDim.x + threadIdx.x;
  if (i < n) p[i] = (bf16)0.0f;
}

// Gather premise embeddings (with premises[:,0]=labels), emit (T*B, E) bf16,
// row m = t*B + b  (time-major for the scan).
__global__ void k_embed_enc(const int* __restrict__ premises, const int* __restrict__ labels,
                            const float* __restrict__ embed, bf16* __restrict__ out) {
  int idx = blockIdx.x * blockDim.x + threadIdx.x;      // TSEQ*BATCH*EDIM = 2M
  if (idx >= TSEQ * BATCH * EDIM) return;
  int m = idx >> 9;            // / EDIM
  int e = idx & (EDIM - 1);
  int t = m >> 6;              // / BATCH
  int b = m & (BATCH - 1);
  int tok = (t == 0) ? labels[b] : premises[b * TSEQ + t];
  out[idx] = (bf16)embed[(size_t)tok * EDIM + e];
}

// Gather decoder inputs embed[hypothesis[:, :-1]], rows padded to 4096 (zeros).
__global__ void k_embed_dec(const int* __restrict__ hyp, const float* __restrict__ embed,
                            bf16* __restrict__ out) {
  int idx = blockIdx.x * blockDim.x + threadIdx.x;      // 4096*EDIM
  if (idx >= 4096 * EDIM) return;
  int m = idx >> 9;
  int e = idx & (EDIM - 1);
  int t = m >> 6;
  int b = m & (BATCH - 1);
  float v = 0.0f;
  if (t < TSEQ - 1) {
    int tok = hyp[b * TSEQ + t];
    v = embed[(size_t)tok * EDIM + e];
  }
  out[idx] = (bf16)v;
}

// ------------------------------ WMMA GEMM ------------------------------
// C[M,N] (f32)  (+)=  A[M,K](bf16,row-major) x W[N,K](bf16,row-major)^T  (+ bias)
// 256 threads = 8 waves; block tile 128(M) x 64(N); wave = 16(M) x 64(N).
// K sliced in steps of 32; A/B slices staged in LDS, double-buffered via
// global_load_async_to_lds_b128 so the DMA of slice k+1 overlaps WMMA on k.
// mode 0: row m stored at C + m*ldc + col
// mode 1: row m = t*64+b stored at C + (b*64 + t + 1)*32000 + col  (logits remap)
__global__ __launch_bounds__(256) void k_gemm_bf16(
    const bf16* __restrict__ A, const bf16* __restrict__ W,
    const float* __restrict__ bias, float* __restrict__ C,
    int M, int N, int K, long long ldc, int beta, int mode)
{
  __shared__ __align__(128) bf16 As[2][128 * 32];   // 2 x 8KB
  __shared__ __align__(128) bf16 Bs[2][64 * 32];    // 2 x 4KB

  const int tid    = threadIdx.x;
  const int wave   = tid >> 5;
  const int lane   = tid & 31;
  const int half   = lane >> 4;       // 0 / 1
  const int mi     = lane & 15;
  const int tileMb = blockIdx.y * 128;
  const int tileM  = tileMb + wave * 16;
  const int tileN  = blockIdx.x * 64;

  const unsigned baseA = lds_off_of(&As[0][0]);
  const unsigned baseB = lds_off_of(&Bs[0][0]);

  // Stage one K-slice: B 64x32 (1 chunk/thread), A 128x32 (2 chunks/thread).
  auto stage = [&](int buf, int kk) {
    {
      const int row = tid >> 2, cc = tid & 3;
      async_copy_b128(baseB + (unsigned)(buf * (64 * 32) + row * 32 + cc * 8) * 2u,
                      W + (size_t)(tileN + row) * K + kk + cc * 8);
    }
#pragma unroll
    for (int s = 0; s < 2; ++s) {
      const int c   = tid + s * 256;
      const int row = c >> 2, cc = c & 3;
      async_copy_b128(baseA + (unsigned)(buf * (128 * 32) + row * 32 + cc * 8) * 2u,
                      A + (size_t)(tileMb + row) * K + kk + cc * 8);
    }
  };

  v8f acc[4];
  acc[0] = v8f{}; acc[1] = v8f{}; acc[2] = v8f{}; acc[3] = v8f{};

  stage(0, 0);
  for (int kk = 0; kk < K; kk += 32) {
    const int buf = (kk >> 5) & 1;
    wait_async0();        // slice kk resident in LDS (this wave's copies)
    __syncthreads();      // ...and everyone else's; all waves done reading buf^1
    if (kk + 32 < K) stage(buf ^ 1, kk + 32);   // DMA next slice, overlaps WMMA

    // A fragment, 16x32 bf16: lane holds row mi,
    // K = [half*16, half*16+16) in elems 0..15 (same order as B).
    const bf16* ap = &As[buf][(wave * 16 + mi) * 32];
    bf16x8 a0 = *(const bf16x8*)(ap + half * 16);
    bf16x8 a1 = *(const bf16x8*)(ap + half * 16 + 8);
    v16bf a;
#pragma unroll
    for (int i = 0; i < 8; ++i) { a[i] = a0[i]; a[i + 8] = a1[i]; }

    // Load ALL four B fragments first (32x16 each: lane = column mi of tile j,
    // K = [half*16, half*16+16)), then run the 4 WMMAs back-to-back behind a
    // single s_wait_dscnt — no inter-WMMA hazards (disjoint accumulators).
    const bf16* bp = &Bs[buf][mi * 32 + half * 16];
    v16bf b0 = *(const v16bf*)(bp);
    v16bf b1 = *(const v16bf*)(bp + 16 * 32);
    v16bf b2 = *(const v16bf*)(bp + 32 * 32);
    v16bf b3 = *(const v16bf*)(bp + 48 * 32);

    acc[0] = __builtin_amdgcn_wmma_f32_16x16x32_bf16(false, a, false, b0, (short)0, acc[0], false, false);
    acc[1] = __builtin_amdgcn_wmma_f32_16x16x32_bf16(false, a, false, b1, (short)0, acc[1], false, false);
    acc[2] = __builtin_amdgcn_wmma_f32_16x16x32_bf16(false, a, false, b2, (short)0, acc[2], false, false);
    acc[3] = __builtin_amdgcn_wmma_f32_16x16x32_bf16(false, a, false, b3, (short)0, acc[3], false, false);
  }

  // C/D layout: element r -> row tileM + half*8 + r, col tileN + j*16 + mi.
#pragma unroll
  for (int j = 0; j < 4; ++j) {
    const int col = tileN + j * 16 + mi;
#pragma unroll
    for (int r = 0; r < 8; ++r) {
      const int row = tileM + half * 8 + r;
      if (row < M) {
        float v = acc[j][r];
        if (bias) v += bias[col];
        float* Cp;
        if (mode == 0) {
          Cp = C + (size_t)row * (size_t)ldc + col;
        } else {
          const int t = row >> 6;           // row = t*64 + b
          const int b = row & 63;
          Cp = C + ((size_t)b * TSEQ + (t + 1)) * (size_t)VDIM + col;
        }
        if (beta) v += *Cp;
        *Cp = v;
      }
    }
  }
}

// ------------------------------ LSTM cell ------------------------------
__global__ void k_lstm_cell(const float* __restrict__ gates,
                            float* __restrict__ c,
                            bf16* __restrict__ h_bf,
                            bf16* __restrict__ c_bf,
                            bf16* __restrict__ hs_row) {
  int idx = blockIdx.x * blockDim.x + threadIdx.x;   // 64*1024
  if (idx >= BATCH * HDIM) return;
  int b = idx >> 10;
  int j = idx & (HDIM - 1);
  const float* g = gates + (size_t)b * (4 * HDIM);
  float gi = g[j];
  float gf = g[HDIM + j];
  float gg = g[2 * HDIM + j];
  float go = g[3 * HDIM + j];
  float si = 1.0f / (1.0f + expf(-gi));
  float sf = 1.0f / (1.0f + expf(-gf));
  float so = 1.0f / (1.0f + expf(-go));
  float cn = sf * c[idx] + si * tanhf(gg);
  float hn = so * tanhf(cn);
  c[idx]    = cn;
  c_bf[idx] = (bf16)cn;
  h_bf[idx] = (bf16)hn;
  if (hs_row) hs_row[idx] = (bf16)hn;
}

// z = mu + eps * exp(0.5*mu); writes f32 and/or bf16 destinations.
__global__ void k_repar_z(const float* __restrict__ mu, const float* __restrict__ eps,
                          float* __restrict__ out_f32, bf16* __restrict__ out_bf) {
  int idx = blockIdx.x * blockDim.x + threadIdx.x;   // 64*1024
  if (idx >= BATCH * HDIM) return;
  float m = mu[idx];
  float z = m + eps[idx] * expf(0.5f * m);
  if (out_f32) out_f32[idx] = z;
  if (out_bf)  out_bf[idx]  = (bf16)z;
}

// raw_outputs[:,0,:] = 0 ; words[:,0] = hypothesis[:,0]
__global__ void k_finalize_head(const int* __restrict__ hyp,
                                float* __restrict__ words, float* __restrict__ raw) {
  int idx = blockIdx.x * blockDim.x + threadIdx.x;   // BATCH * VDIM
  if (idx >= BATCH * VDIM) return;
  int b = idx / VDIM;
  int v = idx - b * VDIM;
  raw[(size_t)b * TSEQ * VDIM + v] = 0.0f;
  if (v == 0) words[b * TSEQ] = (float)hyp[b * TSEQ];
}

// One block per (b, t=1..63): words[b,t] = argmax_v raw[b,t,v] (first max wins).
__global__ __launch_bounds__(256) void k_argmax(const float* __restrict__ raw,
                                                float* __restrict__ words) {
  const int b = blockIdx.x;
  const int t = blockIdx.y + 1;
  const float* p = raw + (size_t)b * TSEQ * VDIM + (size_t)t * VDIM;
  const int tid = threadIdx.x;
  float best = -__builtin_huge_valf();
  int   bi   = 0;
  for (int i = tid; i < VDIM; i += 256) {
    float v = p[i];
    if (v > best) { best = v; bi = i; }
  }
  __shared__ float sv[256];
  __shared__ int   si[256];
  sv[tid] = best; si[tid] = bi;
  __syncthreads();
  for (int s = 128; s > 0; s >>= 1) {
    if (tid < s) {
      if (sv[tid + s] > sv[tid] || (sv[tid + s] == sv[tid] && si[tid + s] < si[tid])) {
        sv[tid] = sv[tid + s]; si[tid] = si[tid + s];
      }
    }
    __syncthreads();
  }
  if (tid == 0) words[b * TSEQ + t] = (float)si[0];
}

// ------------------------------ orchestration ------------------------------

extern "C" void kernel_launch(void* const* d_in, const int* in_sizes, int n_in,
                              void* d_out, int out_size, void* d_ws, size_t ws_size,
                              hipStream_t stream) {
  const int*   premises = (const int*)d_in[0];
  const int*   hyp      = (const int*)d_in[1];
  const int*   labels   = (const int*)d_in[2];
  const float* embed_w  = (const float*)d_in[3];
  const float* enc_Wih  = (const float*)d_in[4];
  const float* enc_Whh  = (const float*)d_in[5];
  const float* enc_bih  = (const float*)d_in[6];
  const float* enc_bhh  = (const float*)d_in[7];
  const float* repar_W  = (const float*)d_in[8];
  const float* repar_b  = (const float*)d_in[9];
  const float* dec_Wih  = (const float*)d_in[10];
  const float* dec_Whh  = (const float*)d_in[11];
  const float* dec_bih  = (const float*)d_in[12];
  const float* dec_bhh  = (const float*)d_in[13];
  const float* out_W    = (const float*)d_in[14];
  const float* out_b    = (const float*)d_in[15];
  const float* eps0     = (const float*)d_in[16];
  const float* eps1     = (const float*)d_in[17];

  float* words = (float*)d_out;
  float* raw   = words + BATCH * TSEQ;   // raw_outputs base (B,T,V)

  // ---- workspace carve (deterministic, 256B aligned) ----
  char*  base = (char*)d_ws;
  size_t off  = 0;
  auto carve = [&](size_t bytes) -> void* {
    void* p = base + off;
    off = (off + bytes + 255) & ~(size_t)255;
    return p;
  };
  bf16*  w_eih  = (bf16*) carve((size_t)4 * HDIM * EDIM * 2);
  bf16*  w_ehh  = (bf16*) carve((size_t)4 * HDIM * HDIM * 2);
  bf16*  w_dih  = (bf16*) carve((size_t)4 * HDIM * EDIM * 2);
  bf16*  w_dhh  = (bf16*) carve((size_t)4 * HDIM * HDIM * 2);
  bf16*  w_rep  = (bf16*) carve((size_t)HDIM * HDIM * 2);
  bf16*  w_out  = (bf16*) carve((size_t)VDIM * HDIM * 2);
  float* bias_e = (float*)carve((size_t)4 * HDIM * 4);
  float* bias_d = (float*)carve((size_t)4 * HDIM * 4);
  bf16*  Aenc   = (bf16*) carve((size_t)4096 * EDIM * 2);
  bf16*  Adec   = (bf16*) carve((size_t)4096 * EDIM * 2);
  float* gates  = (float*)carve((size_t)4096 * 4096 * 4);
  bf16*  h_bf   = (bf16*) carve((size_t)128 * HDIM * 2);   // 128 rows: WMMA M-pad
  bf16*  c_bf   = (bf16*) carve((size_t)128 * HDIM * 2);
  float* c_f    = (float*)carve((size_t)BATCH * HDIM * 4);
  float* mu0    = (float*)carve((size_t)BATCH * HDIM * 4);
  float* mu1    = (float*)carve((size_t)BATCH * HDIM * 4);
  bf16*  hs_bf  = (bf16*) carve((size_t)4096 * HDIM * 2);  // (t*64+b) rows, 4032 used

  const int TB = 256;
  auto blk = [](long long n, int t) { return (unsigned)((n + t - 1) / t); };

  // 1) weights -> bf16 (all bf16 weights ~93MB: resident in 192MB L2)
  k_f32_to_bf16<<<blk((long long)4*HDIM*EDIM, TB), TB, 0, stream>>>(enc_Wih, w_eih, 4*HDIM*EDIM);
  k_f32_to_bf16<<<blk((long long)4*HDIM*HDIM, TB), TB, 0, stream>>>(enc_Whh, w_ehh, 4*HDIM*HDIM);
  k_f32_to_bf16<<<blk((long long)4*HDIM*EDIM, TB), TB, 0, stream>>>(dec_Wih, w_dih, 4*HDIM*EDIM);
  k_f32_to_bf16<<<blk((long long)4*HDIM*HDIM, TB), TB, 0, stream>>>(dec_Whh, w_dhh, 4*HDIM*HDIM);
  k_f32_to_bf16<<<blk((long long)HDIM*HDIM,   TB), TB, 0, stream>>>(repar_W, w_rep, HDIM*HDIM);
  k_f32_to_bf16<<<blk((long long)VDIM*HDIM,   TB), TB, 0, stream>>>(out_W,   w_out, VDIM*HDIM);
  k_bias_sum<<<blk(4*HDIM, TB), TB, 0, stream>>>(enc_bih, enc_bhh, bias_e, 4*HDIM);
  k_bias_sum<<<blk(4*HDIM, TB), TB, 0, stream>>>(dec_bih, dec_bhh, bias_d, 4*HDIM);

  // 2) initial state (ws is poisoned once; re-init every call)
  k_zero_f32 <<<blk(BATCH*HDIM, TB), TB, 0, stream>>>(c_f, BATCH*HDIM);
  k_zero_bf16<<<blk(128*HDIM,   TB), TB, 0, stream>>>(h_bf, 128*HDIM);

  // 3) encoder input GEMM for all timesteps: gates = Aenc @ Wih^T + (bih+bhh)
  k_embed_enc<<<blk((long long)TSEQ*BATCH*EDIM, TB), TB, 0, stream>>>(premises, labels, embed_w, Aenc);
  k_gemm_bf16<<<dim3(4*HDIM/64, 4096/128), TB, 0, stream>>>(
      Aenc, w_eih, bias_e, gates, 4096, 4*HDIM, EDIM, 4*HDIM, 0, 0);

  // 4) encoder recurrence
  for (int t = 0; t < TSEQ; ++t) {
    float* g_t = gates + (size_t)t * BATCH * 4 * HDIM;
    k_gemm_bf16<<<dim3(4*HDIM/64, 1), TB, 0, stream>>>(
        h_bf, w_ehh, nullptr, g_t, BATCH, 4*HDIM, HDIM, 4*HDIM, 1, 0);
    k_lstm_cell<<<blk(BATCH*HDIM, TB), TB, 0, stream>>>(g_t, c_f, h_bf, c_bf, nullptr);
  }

  // 5) reparametrize: mu0 = h_enc @ repar_W^T + b ; mu1 = c_enc @ repar_W^T + b
  k_gemm_bf16<<<dim3(HDIM/64, 1), TB, 0, stream>>>(h_bf, w_rep, repar_b, mu0, BATCH, HDIM, HDIM, HDIM, 0, 0);
  k_gemm_bf16<<<dim3(HDIM/64, 1), TB, 0, stream>>>(c_bf, w_rep, repar_b, mu1, BATCH, HDIM, HDIM, HDIM, 0, 0);
  k_repar_z<<<blk(BATCH*HDIM, TB), TB, 0, stream>>>(mu0, eps0, nullptr, h_bf);  // decoder h0
  k_repar_z<<<blk(BATCH*HDIM, TB), TB, 0, stream>>>(mu1, eps1, c_f, nullptr);   // decoder c0

  // 6) decoder input GEMM for all 63 timesteps (rows padded to 4096 with zeros)
  k_embed_dec<<<blk((long long)4096*EDIM, TB), TB, 0, stream>>>(hyp, embed_w, Adec);
  k_gemm_bf16<<<dim3(4*HDIM/64, 4096/128), TB, 0, stream>>>(
      Adec, w_dih, bias_d, gates, (TSEQ-1)*BATCH, 4*HDIM, EDIM, 4*HDIM, 0, 0);

  // 7) decoder recurrence; h emitted straight into bf16 hs for the output GEMM
  for (int t = 0; t < TSEQ - 1; ++t) {
    float* g_t = gates + (size_t)t * BATCH * 4 * HDIM;
    k_gemm_bf16<<<dim3(4*HDIM/64, 1), TB, 0, stream>>>(
        h_bf, w_dhh, nullptr, g_t, BATCH, 4*HDIM, HDIM, 4*HDIM, 1, 0);
    k_lstm_cell<<<blk(BATCH*HDIM, TB), TB, 0, stream>>>(
        g_t, c_f, h_bf, c_bf, hs_bf + (size_t)t * BATCH * HDIM);
  }

  // 8) output projection: logits = hs @ out_W^T + out_b -> raw[b, t+1, :]
  //    M=4032, N=32000, K=1024 — the 264-GFLOP bulk of the workload.
  k_gemm_bf16<<<dim3(VDIM/64, 4096/128), TB, 0, stream>>>(
      hs_bf, w_out, out_b, raw, (TSEQ-1)*BATCH, VDIM, HDIM, 0, 0, 1);

  // 9) head row + words
  k_finalize_head<<<blk((long long)BATCH*VDIM, TB), TB, 0, stream>>>(hyp, words, raw);
  k_argmax<<<dim3(BATCH, TSEQ-1), TB, 0, stream>>>(raw, words);
}
